// KalmanNetNN_78597901517468
// MI455X (gfx1250) — matrix operand order repaired and checked by reading, not compiled, not run on hardware
//
#include <hip/hip_runtime.h>
#include <hip/hip_bf16.h>

// ---------------------------------------------------------------------------
// KalmanNet on MI455X (gfx1250): memory-bound GEMV chain.
//  * one-time fp32->bf16 weight conversion (halves the per-step weight stream,
//    24.7 GB total -> ~1.05 ms HBM roofline at 23.3 TB/s)
//  * GEMVs via V_WMMA_F32_16X16X32_BF16, x-vector in column 0 of B,
//    16 output rows per wave32
//  * split-K partials (deterministic, no FP atomics) for memory-level
//    parallelism: W_hh GEMV becomes 3840 waves instead of 960
//  * x staged to LDS via GLOBAL_LOAD_ASYNC_TO_LDS_B128 (ASYNCcnt path)
// ---------------------------------------------------------------------------

typedef __bf16 bf16_t;
typedef __attribute__((ext_vector_type(16))) __bf16 v16bf;
typedef __attribute__((ext_vector_type(8)))  __bf16 v8bf;
typedef __attribute__((ext_vector_type(8)))  float  v8f;
typedef __attribute__((ext_vector_type(4)))  int    v4i;

#define KN_M   16
#define KN_N   16
#define KN_T   100
#define KN_H1  2560
#define KN_H2  1024
#define KN_HID 5120
#define KN_KSLICE 1280        // max K elements per wave (split-K slice)

__device__ __forceinline__ bf16_t f2bf(float f) {
    unsigned u = __builtin_bit_cast(unsigned, f);
    u += 0x7FFFu + ((u >> 16) & 1u);          // round-to-nearest-even
    unsigned short s = (unsigned short)(u >> 16);
    return __builtin_bit_cast(bf16_t, s);
}

// Stage n bf16 (n % 8 == 0, 16B-aligned) from global to LDS.
// Uses the gfx1250 async-to-LDS path (ASYNCcnt) when the builtins exist.
// Builtin signature on this toolchain: (v4i AS1* src, v4i AS3* dst, Ii, Ii).
__device__ __forceinline__ void stage_x(bf16_t* sdst, const bf16_t* gsrc, int n) {
#if defined(__gfx1250__) && \
    __has_builtin(__builtin_amdgcn_global_load_async_to_lds_b128) && \
    __has_builtin(__builtin_amdgcn_s_wait_asynccnt)
    for (int i = threadIdx.x * 8; i < n; i += blockDim.x * 8) {
        __builtin_amdgcn_global_load_async_to_lds_b128(
            (__attribute__((address_space(1))) v4i*)(gsrc + i),
            (__attribute__((address_space(3))) v4i*)(sdst + i),
            0, 0);
    }
    __builtin_amdgcn_s_wait_asynccnt(0);
#else
    for (int i = threadIdx.x; i < n; i += blockDim.x) sdst[i] = gsrc[i];
#endif
    __syncthreads();
}

// ---------------------------------------------------------------------------
// fp32 -> bf16 streaming conversion (one-time, grid-stride)
// ---------------------------------------------------------------------------
__global__ void k_f32_to_bf16(const float* __restrict__ src,
                              bf16_t* __restrict__ dst, size_t n) {
    size_t i = (size_t)blockIdx.x * blockDim.x + threadIdx.x;
    size_t stride = (size_t)gridDim.x * blockDim.x;
    for (; i < n; i += stride) dst[i] = f2bf(src[i]);
}

// ---------------------------------------------------------------------------
// init state: post = sp_post = m1_0 ; h = h0 (fp32 + bf16 copies)
// ---------------------------------------------------------------------------
__global__ void k_init(const float* __restrict__ m1_0, const float* __restrict__ h0,
                       float* __restrict__ post, float* __restrict__ sp_post,
                       float* __restrict__ h, bf16_t* __restrict__ hb) {
    int i = blockIdx.x * blockDim.x + threadIdx.x;
    if (i < KN_M) { post[i] = m1_0[i]; sp_post[i] = m1_0[i]; }
    if (i < KN_HID) { h[i] = h0[i]; hb[i] = f2bf(h0[i]); }
}

// ---------------------------------------------------------------------------
// Split-K GEMV via WMMA bf16.
//   W: [rows x K] row-major bf16 (rows%16==0, K%32==0)
//   blockIdx.y = K-slice (slice length Ks = K/NS, Ks <= KN_KSLICE)
//   NS==1: y = act(W@x + bias) written directly (y_f fp32 / y_b bf16)
//   NS>1 : fp32 partial per slice -> part[slice*rows + row], reduced later
//   A-tile: lane m / m+16 holds row m, two b128 loads per 32-K chunk.
//   B: x in every column  =>  b = x[k0+hi8..+7], x[k0+16+hi8..+7] (table k order).
//   C column 0 lives in lanes 0 (M=0..7) and 16 (M=8..15), VGPRs 0..7.
// ---------------------------------------------------------------------------
__global__ void k_gemv_wmma(const bf16_t* __restrict__ W,
                            const float* __restrict__ bias,
                            const bf16_t* __restrict__ x,
                            float* __restrict__ y_f,
                            bf16_t* __restrict__ y_b,
                            float* __restrict__ part,
                            int rows, int K, int Ks, int NS, int relu) {
    __shared__ __align__(16) bf16_t sx[KN_KSLICE];
    const int slice   = blockIdx.y;
    const int k_begin = slice * Ks;
    stage_x(sx, x + k_begin, Ks);

    const int lane  = threadIdx.x & 31;
    const int gwave = (blockIdx.x * blockDim.x + threadIdx.x) >> 5;
    const int row0  = gwave * 16;
    if (row0 >= rows) return;

    const int m   = lane & 15;
    const int hi8 = (lane >> 4) << 3;          // 0 or 8
    const bf16_t* rowp = W + (size_t)(row0 + m) * K + k_begin + hi8;

    v8f c = {};
#pragma unroll 4
    for (int k0 = 0; k0 < Ks; k0 += 32) {
        __builtin_prefetch(rowp + k0 + 512, 0, 0);   // global_prefetch_b8
        union { v16bf v; v8bf h[2]; } a;
        a.h[0] = *(const v8bf*)(rowp + k0);          // K = hi8..hi8+7
        a.h[1] = *(const v8bf*)(rowp + k0 + 16);     // K = 16+hi8..16+hi8+7
        v16bf b = __builtin_shufflevector(*(const v8bf*)(sx + k0 + hi8), *(const v8bf*)(sx + k0 + 16 + hi8),
                                          0, 1, 2, 3, 4, 5, 6, 7, 8, 9, 10, 11, 12, 13, 14, 15);   // every column of B = x, table k order
        c = __builtin_amdgcn_wmma_f32_16x16x32_bf16(
                false, a.v, false, b, (short)0, c, false, false);
    }

    if (lane == 0 || lane == 16) {
        const int base = row0 + ((lane >> 4) << 3);
        if (NS == 1) {
#pragma unroll
            for (int r = 0; r < 8; ++r) {
                float v = c[r] + (bias ? bias[base + r] : 0.0f);
                if (relu) v = fmaxf(v, 0.0f);
                if (y_f) y_f[base + r] = v;
                if (y_b) y_b[base + r] = f2bf(v);
            }
        } else {
            float* p = part + (size_t)slice * rows + base;
#pragma unroll
            for (int r = 0; r < 8; ++r) p[r] = c[r];
        }
    }
}

// Deterministic split-K reduction: y = act(sum_s part[s] + bias)
__global__ void k_gemv_reduce(const float* __restrict__ part,
                              const float* __restrict__ bias,
                              float* __restrict__ y_f, bf16_t* __restrict__ y_b,
                              int rows, int NS, int relu) {
    int i = blockIdx.x * blockDim.x + threadIdx.x;
    if (i >= rows) return;
    float s = bias ? bias[i] : 0.0f;
    for (int sl = 0; sl < NS; ++sl) s += part[(size_t)sl * rows + i];
    if (relu) s = fmaxf(s, 0.0f);
    if (y_f) y_f[i] = s;
    if (y_b) y_b[i] = f2bf(s);
}

// ---------------------------------------------------------------------------
// Per-step Kalman prior + input normalization (single wave).
// ---------------------------------------------------------------------------
__global__ void k_prior(const float* __restrict__ F, const float* __restrict__ Hm,
                        const float* __restrict__ y, int t,
                        const float* __restrict__ post, float* __restrict__ sp_post,
                        float* __restrict__ m1x, float* __restrict__ m1y,
                        bf16_t* __restrict__ kin_b) {
    __shared__ float s_post[16], s_sp[16], s_pr[16], s_sppr[16];
    __shared__ float s_m1y[16], s_d[16], s_yn[16];
    __shared__ float s_nd, s_ny;
    const int l = threadIdx.x;

    if (l < 16) { s_post[l] = post[l]; s_sp[l] = sp_post[l]; }
    __syncthreads();
    if (l < 16) {
        float a = 0.f, b = 0.f;
        for (int k = 0; k < 16; ++k) {
            float f = F[l * 16 + k];
            a += f * s_post[k];
            b += f * s_sp[k];
        }
        s_pr[l] = a; s_sppr[l] = b;
    }
    __syncthreads();
    if (l < 16) {
        float a = 0.f, b = 0.f;
        for (int k = 0; k < 16; ++k) {
            float h = Hm[l * 16 + k];
            a += h * s_pr[k];
            b += h * s_sppr[k];
        }
        s_m1y[l] = a;                         // m1y
        s_d[l]   = s_post[l] - s_sp[l];       // post - sp_post
        s_yn[l]  = y[l * KN_T + t] - b;       // y_t - obs0
    }
    __syncthreads();
    if (l == 0) {
        float nd = 0.f, ny = 0.f;
        for (int k = 0; k < 16; ++k) { nd += s_d[k] * s_d[k]; ny += s_yn[k] * s_yn[k]; }
        s_nd = fmaxf(sqrtf(nd), 1e-12f);
        s_ny = fmaxf(sqrtf(ny), 1e-12f);
    }
    __syncthreads();
    if (l < 16) {
        kin_b[l]      = f2bf(s_d[l]  / s_nd);
        kin_b[16 + l] = f2bf(s_yn[l] / s_ny);
        m1x[l] = s_pr[l];
        m1y[l] = s_m1y[l];
        sp_post[l] = s_sppr[l];               // next step's sp_post = sp_prior
    }
}

// ---------------------------------------------------------------------------
// GRU gate pointwise math; updates h (fp32) and hb (bf16) in place.
// ---------------------------------------------------------------------------
__global__ void k_gru_gates(const float* __restrict__ gi, const float* __restrict__ gh,
                            float* __restrict__ h, bf16_t* __restrict__ hb) {
    int i = blockIdx.x * blockDim.x + threadIdx.x;
    if (i >= KN_HID) return;
    float r  = 1.0f / (1.0f + expf(-(gi[i]            + gh[i])));
    float z  = 1.0f / (1.0f + expf(-(gi[KN_HID + i]   + gh[KN_HID + i])));
    float nn = tanhf(gi[2 * KN_HID + i] + r * gh[2 * KN_HID + i]);
    float hn = (1.0f - z) * nn + z * h[i];
    h[i] = hn;
    hb[i] = f2bf(hn);
}

// ---------------------------------------------------------------------------
// Fused tail: KG = W3@l2 + b3 (16 waves of WMMA), then
// post = m1x + KG @ (y_t - m1y), write out[:, t].  512 threads.
// ---------------------------------------------------------------------------
__global__ void k_final(const bf16_t* __restrict__ W3b, const float* __restrict__ b3,
                        const bf16_t* __restrict__ l2b,
                        const float* __restrict__ m1x, const float* __restrict__ m1y,
                        const float* __restrict__ y, int t,
                        float* __restrict__ post, float* __restrict__ out) {
    __shared__ __align__(16) bf16_t sx[KN_H2];
    __shared__ float kg[256];
    stage_x(sx, l2b, KN_H2);

    const int lane = threadIdx.x & 31;
    const int wid  = threadIdx.x >> 5;         // 0..15
    const int row0 = wid * 16;
    const int K    = KN_H2;
    const int m    = lane & 15;
    const int hi8  = (lane >> 4) << 3;
    const bf16_t* rowp = W3b + (size_t)(row0 + m) * K + hi8;

    v8f c = {};
#pragma unroll 4
    for (int k0 = 0; k0 < K; k0 += 32) {
        union { v16bf v; v8bf h[2]; } a;
        a.h[0] = *(const v8bf*)(rowp + k0);
        a.h[1] = *(const v8bf*)(rowp + k0 + 16);
        v16bf b = __builtin_shufflevector(*(const v8bf*)(sx + k0 + hi8), *(const v8bf*)(sx + k0 + 16 + hi8),
                                          0, 1, 2, 3, 4, 5, 6, 7, 8, 9, 10, 11, 12, 13, 14, 15);
        c = __builtin_amdgcn_wmma_f32_16x16x32_bf16(
                false, a.v, false, b, (short)0, c, false, false);
    }
    if (lane == 0 || lane == 16) {
        const int base = row0 + ((lane >> 4) << 3);
#pragma unroll
        for (int r = 0; r < 8; ++r) kg[base + r] = c[r] + b3[base + r];
    }
    __syncthreads();

    if (threadIdx.x < 16) {
        const int mr = threadIdx.x;
        float s = m1x[mr];
        for (int n = 0; n < 16; ++n) {
            float dy = y[n * KN_T + t] - m1y[n];
            s += kg[mr * 16 + n] * dy;
        }
        post[mr] = s;
        out[mr * KN_T + t] = s;
    }
}

// ---------------------------------------------------------------------------
// host side
// ---------------------------------------------------------------------------
static inline size_t align256(size_t x) { return (x + 255) & ~(size_t)255; }

extern "C" void kernel_launch(void* const* d_in, const int* in_sizes, int n_in,
                              void* d_out, int out_size, void* d_ws, size_t ws_size,
                              hipStream_t stream) {
    (void)in_sizes; (void)n_in; (void)out_size;

    const float* y    = (const float*)d_in[0];
    const float* F    = (const float*)d_in[1];
    const float* Hm   = (const float*)d_in[2];
    const float* m1_0 = (const float*)d_in[3];
    const float* h0   = (const float*)d_in[4];
    const float* W1   = (const float*)d_in[5];
    const float* b1   = (const float*)d_in[6];
    const float* W_ih = (const float*)d_in[7];
    const float* b_ih = (const float*)d_in[8];
    const float* W_hh = (const float*)d_in[9];
    const float* b_hh = (const float*)d_in[10];
    const float* W2   = (const float*)d_in[11];
    const float* b2   = (const float*)d_in[12];
    const float* W3   = (const float*)d_in[13];
    const float* b3   = (const float*)d_in[14];
    float* out = (float*)d_out;

    // ---- workspace layout (bf16 weights + state + activations) ----
    char* base = (char*)d_ws;
    size_t off = 0;
    auto take = [&](size_t bytes) -> void* {
        void* p = base + off;
        off = align256(off + bytes);
        return p;
    };
    const size_t nW1  = (size_t)KN_H1 * (KN_M + KN_N);          //  81,920
    const size_t nWih = (size_t)3 * KN_HID * KN_H1;             //  39,321,600
    const size_t nWhh = (size_t)3 * KN_HID * KN_HID;            //  78,643,200
    const size_t nW2  = (size_t)KN_H2 * KN_HID;                 //   5,242,880
    const size_t nW3  = (size_t)(KN_M * KN_N) * KN_H2;          //     262,144

    bf16_t* W1b  = (bf16_t*)take(nW1  * sizeof(bf16_t));
    bf16_t* Wihb = (bf16_t*)take(nWih * sizeof(bf16_t));
    bf16_t* Whhb = (bf16_t*)take(nWhh * sizeof(bf16_t));
    bf16_t* W2b  = (bf16_t*)take(nW2  * sizeof(bf16_t));
    bf16_t* W3b  = (bf16_t*)take(nW3  * sizeof(bf16_t));

    float*  post = (float*)take(16 * sizeof(float));
    float*  spp  = (float*)take(16 * sizeof(float));
    float*  m1x  = (float*)take(16 * sizeof(float));
    float*  m1y  = (float*)take(16 * sizeof(float));
    bf16_t* kinb = (bf16_t*)take(32 * sizeof(bf16_t));
    bf16_t* l1b  = (bf16_t*)take((size_t)KN_H1 * sizeof(bf16_t));
    float*  gi   = (float*)take((size_t)3 * KN_HID * sizeof(float));
    float*  gh   = (float*)take((size_t)3 * KN_HID * sizeof(float));
    float*  h    = (float*)take((size_t)KN_HID * sizeof(float));
    bf16_t* hb   = (bf16_t*)take((size_t)KN_HID * sizeof(bf16_t));
    bf16_t* l2b  = (bf16_t*)take((size_t)KN_H2 * sizeof(bf16_t));
    float*  part = (float*)take((size_t)4 * 3 * KN_HID * sizeof(float)); // split-K partials

    if (off > ws_size) return;   // workspace too small: refuse to scribble OOB

    // ---- one-time weight conversion fp32 -> bf16 (streamed, ~740 MB total) ----
    k_f32_to_bf16<<<1024, 256, 0, stream>>>(W1,   W1b,  nW1);
    k_f32_to_bf16<<<4096, 256, 0, stream>>>(W_ih, Wihb, nWih);
    k_f32_to_bf16<<<4096, 256, 0, stream>>>(W_hh, Whhb, nWhh);
    k_f32_to_bf16<<<2048, 256, 0, stream>>>(W2,   W2b,  nW2);
    k_f32_to_bf16<<<1024, 256, 0, stream>>>(W3,   W3b,  nW3);

    k_init<<<(KN_HID + 255) / 256, 256, 0, stream>>>(m1_0, h0, post, spp, h, hb);

    auto gemv = [&](const bf16_t* W, const float* bias, const bf16_t* x,
                    float* yf, bf16_t* yb, int rows, int K, int relu) {
        int NS = K / KN_KSLICE; if (NS < 1) NS = 1;
        int Ks = K / NS;
        dim3 grid((rows / 16 + 3) / 4, NS);    // 4 waves (128 threads) per block
        if (NS == 1) {
            k_gemv_wmma<<<grid, 128, 0, stream>>>(W, bias, x, yf, yb, nullptr,
                                                  rows, K, Ks, NS, relu);
        } else {
            k_gemv_wmma<<<grid, 128, 0, stream>>>(W, nullptr, x, nullptr, nullptr,
                                                  part, rows, K, Ks, NS, relu);
            k_gemv_reduce<<<(rows + 255) / 256, 256, 0, stream>>>(
                part, bias, yf, yb, rows, NS, relu);
        }
    };

    for (int t = 0; t < KN_T; ++t) {
        k_prior<<<1, 32, 0, stream>>>(F, Hm, y, t, post, spp, m1x, m1y, kinb);
        // l1 = relu(W1 @ kin + b1)                 [2560], K=32 (one WMMA/tile)
        gemv(W1b, b1, kinb, nullptr, l1b, KN_H1, KN_M + KN_N, 1);
        // gi = W_ih @ l1 + b_ih                    [15360], K=2560, split-K=2
        gemv(Wihb, b_ih, l1b, gi, nullptr, 3 * KN_HID, KN_H1, 0);
        // gh = W_hh @ h + b_hh                     [15360], K=5120, split-K=4
        gemv(Whhb, b_hh, hb, gh, nullptr, 3 * KN_HID, KN_HID, 0);
        // GRU gates -> h, hb (in place)
        k_gru_gates<<<KN_HID / 256, 256, 0, stream>>>(gi, gh, h, hb);
        // l2 = relu(W2 @ h_new + b2)               [1024], K=5120, split-K=4
        gemv(W2b, b2, hb, nullptr, l2b, KN_H2, KN_HID, 1);
        // KG = W3 @ l2 + b3 ; post = m1x + KG @ dy ; out[:, t]
        k_final<<<1, 512, 0, stream>>>(W3b, b3, l2b, m1x, m1y, y, t, post, out);
    }
}
